// EdgeMLPMixerPolicy_52793738003056
// MI455X (gfx1250) — compile-verified, matrix operand order checked by reading, not run on hardware
//
#include <hip/hip_runtime.h>
#include <hip/hip_bf16.h>

#define HDIM 256
#define H2   512
#define GNUM 2048
#define ENUM 262144
#define TM   16

typedef __attribute__((ext_vector_type(16))) _Float16 v16h;
typedef __attribute__((ext_vector_type(8)))  float    v8f;

__device__ __forceinline__ float gelu_f(float x) {
  return 0.5f * x * (1.0f + erff(x * 0.7071067811865475f));
}

// sum across a 16-lane group (wave32: two groups per wave)
__device__ __forceinline__ float redsum16(float v) {
  v += __shfl_xor(v, 1);
  v += __shfl_xor(v, 2);
  v += __shfl_xor(v, 4);
  v += __shfl_xor(v, 8);
  return v;
}

// 16-bit A-matrix 16xK fragment per CDNA5 ISA layout (16x32 per step).
// Per lane this is two contiguous 16-byte runs -> 2x ds_load_b128.
__device__ __forceinline__ v16h fragA(const _Float16* A, int lane, int k0, int ldk) {
  const int m  = lane & 15;
  const int kh = (lane >> 4) << 3;
  v16h a;
#pragma unroll
  for (int j = 0; j < 8; ++j) {
    const int kk = k0 + ((j >> 2) << 4) + kh + ((j & 3) << 1);
    a[2 * j]     = A[m * ldk + kk];
    a[2 * j + 1] = A[m * ldk + kk + 1];
  }
  return a;
}

// B fragment from pre-permuted weights: one 32B contiguous load per lane.
// Packed layout: (((nt*(K/32)+kb)*32 + lane)*16 + idx)
__device__ __forceinline__ v16h fragBpk(const _Float16* Wpk, int lane, int kb, int nt,
                                        int kblocks) {
  const size_t off = (((size_t)(nt * kblocks + kb) * 32) + lane) << 4;
  return *(const v16h*)(Wpk + off);
}

// Source index in row-major [K][256] for packed element i (see fragB layout).
__device__ __forceinline__ int wpk_src_index(int i, int K) {
  const int idx  = i & 15;
  const int lane = (i >> 4) & 31;
  const int kblk = i >> 9;            // nt*(K/32) + kb
  const int kb   = kblk % (K / 32);
  const int nt   = kblk / (K / 32);
  const int j    = idx >> 1;
  const int kk   = kb * 32 + ((j >> 2) << 4) + ((lane >> 4) << 3) + ((j & 3) << 1) + (idx & 1);
  const int n    = nt * 16 + (lane & 15);
  return kk * HDIM + n;
}

// [16 x KTOT] (LDS f16) @ [KTOT x 256] (packed f16 weights) -> gelu(acc+bias) into Out
// 8 waves x 2 N-tiles each; EXEC uniform (no divergence around WMMA).
template <int KTOT, bool ADDTO>
__device__ __forceinline__ void wmma_rowtile(const _Float16* A, const _Float16* Wpk,
                                             const float* bias, float* Out,
                                             int wave, int lane) {
  constexpr int KB = KTOT / 32;
#pragma unroll
  for (int t = 0; t < 2; ++t) {
    const int nt = wave * 2 + t;
    v8f acc = {};
#pragma unroll
    for (int kb = 0; kb < KB; ++kb) {
      v16h a = fragA(A, lane, kb * 32, KTOT);
      v16h b = fragBpk(Wpk, lane, kb, nt, KB);
      acc = __builtin_amdgcn_wmma_f32_16x16x32_f16(false, a, false, b, (short)0, acc,
                                                   false, false);
    }
    const int n  = nt * 16 + (lane & 15);
    const int mb = (lane >> 4) << 3;
    const float bn = bias[n];
#pragma unroll
    for (int rr = 0; rr < 8; ++rr) {
      const float v = gelu_f(acc[rr] + bn);
      if (ADDTO) Out[(mb + rr) * HDIM + n] += v;
      else       Out[(mb + rr) * HDIM + n]  = v;
    }
  }
}

// LayerNorm one 256-row handled by a 16-thread group; output f16 (WMMA A operand)
__device__ __forceinline__ void ln_row_to_f16(const float* in, const float* addv, int cg,
                                              const float* g, const float* bp,
                                              _Float16* out) {
  float s = 0.f, s2 = 0.f;
#pragma unroll
  for (int i = 0; i < 16; ++i) {
    const int c = cg * 16 + i;
    float v = in[c];
    if (addv) v += addv[c];
    s += v; s2 += v * v;
  }
  s = redsum16(s); s2 = redsum16(s2);
  const float mean = s * (1.0f / HDIM);
  const float rs   = rsqrtf(s2 * (1.0f / HDIM) - mean * mean + 1e-5f);
#pragma unroll
  for (int i = 0; i < 16; ++i) {
    const int c = cg * 16 + i;
    float v = in[c];
    if (addv) v += addv[c];
    out[c] = (_Float16)((v - mean) * rs * g[c] + bp[c]);
  }
}

// ---------------------------------------------------------------------------
// Kernel 0: zero scatter buffers; permute+convert weights fp32 -> packed f16
// ---------------------------------------------------------------------------
__global__ __launch_bounds__(256) void k_prep(
    const float* __restrict__ qfW, const float* __restrict__ mlpW,
    const float* __restrict__ lhW1, const float* __restrict__ spW1,
    _Float16* __restrict__ qfWpk, _Float16* __restrict__ mlpWpk,
    _Float16* __restrict__ lhWpk, _Float16* __restrict__ spWpk,
    float* __restrict__ selsum, float* __restrict__ cnt) {
  const int i = blockIdx.x * blockDim.x + threadIdx.x;
  if (i < HDIM * HDIM) qfWpk[i] = (_Float16)qfW[wpk_src_index(i, HDIM)];
  if (i < 2 * HDIM * HDIM) {
    const int l = i / (HDIM * HDIM);
    const int r = i - l * (HDIM * HDIM);
    mlpWpk[i] = (_Float16)mlpW[l * HDIM * HDIM + wpk_src_index(r, HDIM)];
  }
  if (i < H2 * HDIM) {
    const int s = wpk_src_index(i, H2);
    lhWpk[i] = (_Float16)lhW1[s];
    spWpk[i] = (_Float16)spW1[s];
  }
  if (i < GNUM * HDIM) selsum[i] = 0.0f;
  if (i < GNUM)        cnt[i] = 0.0f;
}

// ---------------------------------------------------------------------------
// Kernel 1: fused per-edge pipeline (order emb + q_film + 2-layer MLP) +
//           masked scatter-add into sel_sum/cnt; spill x to workspace.
// ---------------------------------------------------------------------------
__global__ __launch_bounds__(256) void k_edge_mlp(
    const float* __restrict__ edge_tokens, const float* __restrict__ question_tokens,
    const float* __restrict__ order_emb, const float* __restrict__ type_emb,
    const float* __restrict__ qf_g, const float* __restrict__ qf_b,
    const float* __restrict__ qf_bias,
    const float* __restrict__ mlp_g, const float* __restrict__ mlp_b,
    const float* __restrict__ mlp_bias,
    const _Float16* __restrict__ qfWpk, const _Float16* __restrict__ mlpWpk,
    const int* __restrict__ edge_batch, const unsigned char* __restrict__ sel_mask,
    const int* __restrict__ sel_order,
    float* __restrict__ selsum, float* __restrict__ cnt, float* __restrict__ xbuf) {
  __shared__ float    sE[TM][HDIM];   // edge accumulator (et, then layer0 out)
  __shared__ float    sX[TM][HDIM];   // q_e, then final x
  __shared__ _Float16 sA[TM][HDIM];   // WMMA A operand (post-LN, f16)

  const int tid  = threadIdx.x;
  const int lane = tid & 31;
  const int wave = tid >> 5;
  const int r    = tid >> 4;
  const int cg   = tid & 15;
  const int e    = blockIdx.x * TM + r;
  const int g    = edge_batch[e];
  int so = sel_order[e];
  so = so < -1 ? -1 : (so > 10 ? 10 : so);
  const int oidx = so + 1;

#pragma unroll
  for (int i = 0; i < 16; ++i) {
    const int c = cg * 16 + i;
    sE[r][c] = edge_tokens[(size_t)e * HDIM + c] + order_emb[oidx * HDIM + c];
    sX[r][c] = question_tokens[(size_t)g * HDIM + c];
  }
  __syncthreads();

  // q_film: et += gelu(LN(q_e) @ qf_W + qf_bias)
  ln_row_to_f16(&sX[r][0], nullptr, cg, qf_g, qf_b, &sA[r][0]);
  __syncthreads();
  wmma_rowtile<HDIM, true>(&sA[0][0], qfWpk, qf_bias, &sE[0][0], wave, lane);
  __syncthreads();

  // MLP layer 0 on x = et + type_emb[2]
  ln_row_to_f16(&sE[r][0], type_emb + 2 * HDIM, cg, mlp_g, mlp_b, &sA[r][0]);
  __syncthreads();
  wmma_rowtile<HDIM, false>(&sA[0][0], mlpWpk, mlp_bias, &sE[0][0], wave, lane);
  __syncthreads();

  // MLP layer 1
  ln_row_to_f16(&sE[r][0], nullptr, cg, mlp_g + HDIM, mlp_b + HDIM, &sA[r][0]);
  __syncthreads();
  wmma_rowtile<HDIM, false>(&sA[0][0], mlpWpk + HDIM * HDIM, mlp_bias + HDIM,
                            &sX[0][0], wave, lane);
  __syncthreads();

  // spill x; masked scatter-add (L2-resident atomics)
  const bool m = sel_mask[e] != 0;
#pragma unroll
  for (int i = 0; i < 16; ++i) {
    const int c = cg * 16 + i;
    const float xv = sX[r][c];
    xbuf[(size_t)e * HDIM + c] = xv;
    if (m) atomicAdd(&selsum[(size_t)g * HDIM + c], xv);
  }
  if (m && cg == 0) atomicAdd(&cnt[g], 1.0f);
}

// ---------------------------------------------------------------------------
// Kernel 2: per-graph: current = LN(sel_sum/cnt + q); stop head (K=512 GEMM)
// ---------------------------------------------------------------------------
__global__ __launch_bounds__(256) void k_graph_head(
    const float* __restrict__ question_tokens,
    const float* __restrict__ selsum, const float* __restrict__ cnt,
    const float* __restrict__ gn_g, const float* __restrict__ gn_b,
    const float* __restrict__ sp_g, const float* __restrict__ sp_b,
    const _Float16* __restrict__ spWpk, const float* __restrict__ sp_b1,
    const float* __restrict__ sp_W2, const float* __restrict__ sp_b2,
    float* __restrict__ stop_out, float* __restrict__ cur_out) {
  __shared__ float    sC[TM][HDIM];
  __shared__ float    sQ[TM][HDIM];
  __shared__ _Float16 sY[TM][H2];
  __shared__ float    sH[TM][HDIM];

  const int tid  = threadIdx.x;
  const int lane = tid & 31;
  const int wave = tid >> 5;
  const int r    = tid >> 4;
  const int cg   = tid & 15;
  const int g    = blockIdx.x * TM + r;
  const float cn = fmaxf(cnt[g], 1.0f);

#pragma unroll
  for (int i = 0; i < 16; ++i) {
    const int c = cg * 16 + i;
    const float q = question_tokens[(size_t)g * HDIM + c];
    sQ[r][c] = q;
    sC[r][c] = selsum[(size_t)g * HDIM + c] / cn + q;
  }
  __syncthreads();

  // current = LN(sC) with gn params (f32, in place); also an output
  {
    float s = 0.f, s2 = 0.f;
#pragma unroll
    for (int i = 0; i < 16; ++i) {
      const float v = sC[r][cg * 16 + i];
      s += v; s2 += v * v;
    }
    s = redsum16(s); s2 = redsum16(s2);
    const float mean = s * (1.0f / HDIM);
    const float rs   = rsqrtf(s2 * (1.0f / HDIM) - mean * mean + 1e-5f);
#pragma unroll
    for (int i = 0; i < 16; ++i) {
      const int c = cg * 16 + i;
      const float cu = (sC[r][c] - mean) * rs * gn_g[c] + gn_b[c];
      sC[r][c] = cu;
      cur_out[(size_t)g * HDIM + c] = cu;
    }
  }
  __syncthreads();

  // LN over concat(current, q) [512] with sp params -> f16
  {
    const float* src = (cg < 8) ? &sC[r][0] : &sQ[r][0];
    const int base = (cg & 7) * 32;
    float s = 0.f, s2 = 0.f;
#pragma unroll
    for (int i = 0; i < 32; ++i) { const float v = src[base + i]; s += v; s2 += v * v; }
    s = redsum16(s); s2 = redsum16(s2);
    const float mean = s * (1.0f / H2);
    const float rs   = rsqrtf(s2 * (1.0f / H2) - mean * mean + 1e-5f);
#pragma unroll
    for (int i = 0; i < 32; ++i) {
      const int c = cg * 32 + i;
      sY[r][c] = (_Float16)((src[base + i] - mean) * rs * sp_g[c] + sp_b[c]);
    }
  }
  __syncthreads();

  wmma_rowtile<H2, false>(&sY[0][0], spWpk, sp_b1, &sH[0][0], wave, lane);
  __syncthreads();

  float p = 0.f;
#pragma unroll
  for (int i = 0; i < 16; ++i) { const int c = cg * 16 + i; p += sH[r][c] * sp_W2[c]; }
  p = redsum16(p);
  if (cg == 0) stop_out[g] = p + sp_b2[0];
}

// ---------------------------------------------------------------------------
// Kernel 3: per-edge lookahead head: nxt LN, concat LN (512), GEMM, dot
// ---------------------------------------------------------------------------
__global__ __launch_bounds__(256) void k_edge_head(
    const float* __restrict__ question_tokens,
    const float* __restrict__ selsum, const float* __restrict__ cnt,
    const float* __restrict__ cur, const float* __restrict__ xbuf,
    const int* __restrict__ edge_batch,
    const float* __restrict__ gn_g, const float* __restrict__ gn_b,
    const float* __restrict__ lh_g, const float* __restrict__ lh_b,
    const _Float16* __restrict__ lhWpk, const float* __restrict__ lh_b1,
    const float* __restrict__ lh_W2, const float* __restrict__ lh_b2,
    float* __restrict__ elog_out) {
  __shared__ float    sCur[TM][HDIM];
  __shared__ float    sN[TM][HDIM];
  __shared__ _Float16 sY[TM][H2];
  __shared__ float    sH[TM][HDIM];

  const int tid  = threadIdx.x;
  const int lane = tid & 31;
  const int wave = tid >> 5;
  const int r    = tid >> 4;
  const int cg   = tid & 15;
  const int e    = blockIdx.x * TM + r;
  const int g    = edge_batch[e];
  const float cn1 = cnt[g] + 1.0f;

#pragma unroll
  for (int i = 0; i < 16; ++i) {
    const int c = cg * 16 + i;
    sCur[r][c] = cur[(size_t)g * HDIM + c];
    sN[r][c] = (selsum[(size_t)g * HDIM + c] + xbuf[(size_t)e * HDIM + c]) / cn1 +
               question_tokens[(size_t)g * HDIM + c];
  }
  __syncthreads();

  // nxt = LN(sN) with gn params (in place)
  {
    float s = 0.f, s2 = 0.f;
#pragma unroll
    for (int i = 0; i < 16; ++i) {
      const float v = sN[r][cg * 16 + i];
      s += v; s2 += v * v;
    }
    s = redsum16(s); s2 = redsum16(s2);
    const float mean = s * (1.0f / HDIM);
    const float rs   = rsqrtf(s2 * (1.0f / HDIM) - mean * mean + 1e-5f);
#pragma unroll
    for (int i = 0; i < 16; ++i) {
      const int c = cg * 16 + i;
      sN[r][c] = (sN[r][c] - mean) * rs * gn_g[c] + gn_b[c];
    }
  }
  __syncthreads();

  // LN over concat(current[eb], nxt) [512] with lh params -> f16
  {
    const float* src = (cg < 8) ? &sCur[r][0] : &sN[r][0];
    const int base = (cg & 7) * 32;
    float s = 0.f, s2 = 0.f;
#pragma unroll
    for (int i = 0; i < 32; ++i) { const float v = src[base + i]; s += v; s2 += v * v; }
    s = redsum16(s); s2 = redsum16(s2);
    const float mean = s * (1.0f / H2);
    const float rs   = rsqrtf(s2 * (1.0f / H2) - mean * mean + 1e-5f);
#pragma unroll
    for (int i = 0; i < 32; ++i) {
      const int c = cg * 32 + i;
      sY[r][c] = (_Float16)((src[base + i] - mean) * rs * lh_g[c] + lh_b[c]);
    }
  }
  __syncthreads();

  wmma_rowtile<H2, false>(&sY[0][0], lhWpk, lh_b1, &sH[0][0], wave, lane);
  __syncthreads();

  float p = 0.f;
#pragma unroll
  for (int i = 0; i < 16; ++i) { const int c = cg * 16 + i; p += sH[r][c] * lh_W2[c]; }
  p = redsum16(p);
  if (cg == 0) elog_out[e] = p + lh_b2[0];
}

// ---------------------------------------------------------------------------
extern "C" void kernel_launch(void* const* d_in, const int* in_sizes, int n_in,
                              void* d_out, int out_size, void* d_ws, size_t ws_size,
                              hipStream_t stream) {
  (void)in_sizes; (void)n_in; (void)out_size; (void)ws_size;
  const float* edge_tokens     = (const float*)d_in[0];
  const float* question_tokens = (const float*)d_in[1];
  const float* order_emb       = (const float*)d_in[2];
  const float* type_emb        = (const float*)d_in[3];
  const float* gn_g            = (const float*)d_in[4];
  const float* gn_b            = (const float*)d_in[5];
  const float* qf_g            = (const float*)d_in[6];
  const float* qf_b            = (const float*)d_in[7];
  const float* qf_W            = (const float*)d_in[8];
  const float* qf_bias         = (const float*)d_in[9];
  const float* mlp_g           = (const float*)d_in[10];
  const float* mlp_b           = (const float*)d_in[11];
  const float* mlp_W           = (const float*)d_in[12];
  const float* mlp_bias        = (const float*)d_in[13];
  const float* lh_g            = (const float*)d_in[14];
  const float* lh_b            = (const float*)d_in[15];
  const float* lh_W1           = (const float*)d_in[16];
  const float* lh_b1           = (const float*)d_in[17];
  const float* lh_W2           = (const float*)d_in[18];
  const float* lh_b2           = (const float*)d_in[19];
  const float* sp_g            = (const float*)d_in[20];
  const float* sp_b            = (const float*)d_in[21];
  const float* sp_W1           = (const float*)d_in[22];
  const float* sp_b1           = (const float*)d_in[23];
  const float* sp_W2           = (const float*)d_in[24];
  const float* sp_b2           = (const float*)d_in[25];
  const int* edge_batch        = (const int*)d_in[26];
  const unsigned char* sel_mask = (const unsigned char*)d_in[27];
  const int* sel_order         = (const int*)d_in[28];

  char* ws = (char*)d_ws;
  size_t off = 0;
  float* selsum = (float*)(ws + off); off += (size_t)GNUM * HDIM * 4;
  float* cnt    = (float*)(ws + off); off += (size_t)GNUM * 4;
  off = (off + 255) & ~(size_t)255;
  _Float16* qfWpk  = (_Float16*)(ws + off); off += (size_t)HDIM * HDIM * 2;
  _Float16* mlpWpk = (_Float16*)(ws + off); off += (size_t)2 * HDIM * HDIM * 2;
  _Float16* lhWpk  = (_Float16*)(ws + off); off += (size_t)H2 * HDIM * 2;
  _Float16* spWpk  = (_Float16*)(ws + off); off += (size_t)H2 * HDIM * 2;
  off = (off + 255) & ~(size_t)255;
  float* xbuf = (float*)(ws + off);

  float* out     = (float*)d_out;
  float* elog    = out;                 // [E]
  float* stoplog = out + ENUM;          // [G]
  float* curout  = out + ENUM + GNUM;   // [G, H]

  k_prep<<<(GNUM * HDIM) / 256, 256, 0, stream>>>(
      qf_W, mlp_W, lh_W1, sp_W1, qfWpk, mlpWpk, lhWpk, spWpk, selsum, cnt);

  k_edge_mlp<<<ENUM / TM, 256, 0, stream>>>(
      edge_tokens, question_tokens, order_emb, type_emb,
      qf_g, qf_b, qf_bias, mlp_g, mlp_b, mlp_bias,
      qfWpk, mlpWpk, edge_batch, sel_mask, sel_order, selsum, cnt, xbuf);

  k_graph_head<<<GNUM / TM, 256, 0, stream>>>(
      question_tokens, selsum, cnt, gn_g, gn_b,
      sp_g, sp_b, spWpk, sp_b1, sp_W2, sp_b2, stoplog, curout);

  k_edge_head<<<ENUM / TM, 256, 0, stream>>>(
      question_tokens, selsum, cnt, curout, xbuf, edge_batch,
      gn_g, gn_b, lh_g, lh_b, lhWpk, lh_b1, lh_W2, lh_b2, elog);
}
